// GATConv_60078002536571
// MI455X (gfx1250) — matrix operand order repaired and checked by reading, not compiled, not run on hardware
//
#include <hip/hip_runtime.h>

typedef __attribute__((ext_vector_type(16))) _Float16 v16h;
typedef __attribute__((ext_vector_type(4)))  _Float16 v4h;
typedef __attribute__((ext_vector_type(8)))  float    v8f;

#define HEADS      4
#define HDIM       32
#define FEATD      128          // HEADS*HDIM == IN_DIM
#define NEG_SLOPE  0.2f

// ---------------------------------------------------------------------------
// 0) f32 -> (f16 hi, f16 lo) split, 4 elements/thread (Markidis split).
//    Done ONCE so the WMMA kernel is pure b128-load + v_wmma.
// ---------------------------------------------------------------------------
__global__ __launch_bounds__(256) void cvt_hilo(
    const float* __restrict__ in, _Float16* __restrict__ hi,
    _Float16* __restrict__ lo, long n)
{
    const long i = ((long)blockIdx.x * blockDim.x + threadIdx.x) * 4;
    if (i >= n) return;
    const float4 f = *(const float4*)(in + i);
    v4h h, l;
    h[0] = (_Float16)f.x; l[0] = (_Float16)(f.x - (float)h[0]);
    h[1] = (_Float16)f.y; l[1] = (_Float16)(f.y - (float)h[1]);
    h[2] = (_Float16)f.z; l[2] = (_Float16)(f.z - (float)h[2]);
    h[3] = (_Float16)f.w; l[3] = (_Float16)(f.w - (float)h[3]);
    *(v4h*)(hi + i) = h;
    *(v4h*)(lo + i) = l;
}

// ---------------------------------------------------------------------------
// 1) feat = x @ W^T via v_wmma_f32_16x16x32_f16, hi/lo compensated (~f32 exact).
//    Block = 256 threads = 8 waves; block covers 16 node-rows, wave w covers
//    output columns [16w, 16w+16). Operand halves are contiguous 16B chunks
//    -> global_load_b128 feeds the WMMA directly.
// ---------------------------------------------------------------------------
__global__ __launch_bounds__(256) void gat_proj_wmma(
    const _Float16* __restrict__ xh, const _Float16* __restrict__ xl,
    const _Float16* __restrict__ wh, const _Float16* __restrict__ wl,
    float* __restrict__ feat, int n_nodes)
{
    const int lane = threadIdx.x & 31;
    const int wave = threadIdx.x >> 5;
    const int m0   = blockIdx.x << 4;     // node-row tile base
    const int n0   = wave << 4;           // output-col tile base
    const int mr   = lane & 15;
    const int kh   = lane >> 4;           // K half-group held by this lane

    int arow = m0 + mr;
    if (arow >= n_nodes) arow = n_nodes - 1;                 // tail-safe clamp
    const _Float16* __restrict__ xhr = xh + (size_t)arow * FEATD;
    const _Float16* __restrict__ xlr = xl + (size_t)arow * FEATD;
    const _Float16* __restrict__ whr = wh + (size_t)(n0 + mr) * FEATD; // B col = lane&15
    const _Float16* __restrict__ wlr = wl + (size_t)(n0 + mr) * FEATD;

    v8f acc = (v8f)0.0f;
    for (int k0 = 0; k0 < FEATD; k0 += 32) {
        v16h a_hi, a_lo, b_hi, b_lo;
        // A (16x32 f16): VGPR v, half kh -> K = 16*(v/4) + 8*kh + 2*(v%4) (+0,1)
        // halves v=0..3 and v=4..7 are each 8 contiguous f16 = 16B
#pragma unroll
        for (int v = 0; v < 8; ++v) {
            const int ka = k0 + ((v >> 2) << 4) + (kh << 3) + ((v & 3) << 1);
            a_hi[2 * v + 0] = xhr[ka + 0];
            a_hi[2 * v + 1] = xhr[ka + 1];
            a_lo[2 * v + 0] = xlr[ka + 0];
            a_lo[2 * v + 1] = xlr[ka + 1];
        }
        // B (32x16 f16): VGPR v, half kh -> K = 16*(v/4) + 8*kh + 2*(v%4) (+0,1)
        // halves v=0..3 and v=4..7 are each 8 contiguous f16 = 16B
#pragma unroll
        for (int v = 0; v < 8; ++v) {
            const int kb = k0 + ((v >> 2) << 4) + (kh << 3) + ((v & 3) << 1);
            b_hi[2 * v + 0] = whr[kb + 0];
            b_hi[2 * v + 1] = whr[kb + 1];
            b_lo[2 * v + 0] = wlr[kb + 0];
            b_lo[2 * v + 1] = wlr[kb + 1];
        }
        acc = __builtin_amdgcn_wmma_f32_16x16x32_f16(false, a_hi, false, b_hi, (short)0, acc, false, false);
        acc = __builtin_amdgcn_wmma_f32_16x16x32_f16(false, a_lo, false, b_hi, (short)0, acc, false, false);
        acc = __builtin_amdgcn_wmma_f32_16x16x32_f16(false, a_hi, false, b_lo, (short)0, acc, false, false);
    }

    // C/D 16x16 f32: lane col = lane&15, VGPR r -> row r + 8*(lane/16)
    const int col = lane & 15;
    float* __restrict__ frow = feat + (size_t)(m0 + (kh << 3)) * FEATD + n0 + col;
    if (m0 + 16 <= n_nodes) {
        // full tile: unguarded coalesced stores
#pragma unroll
        for (int r = 0; r < 8; ++r)
            frow[(size_t)r * FEATD] = acc[r];
    } else {
#pragma unroll
        for (int r = 0; r < 8; ++r) {
            if (m0 + r + (kh << 3) < n_nodes)
                frow[(size_t)r * FEATD] = acc[r];
        }
    }
}

// ---------------------------------------------------------------------------
// 2) el[n,h] = <feat[n,h,:], attn_l[h,:]> ; er likewise. One thread per (n,h).
// ---------------------------------------------------------------------------
__global__ __launch_bounds__(256) void gat_el_er(
    const float* __restrict__ feat,
    const float* __restrict__ attn_l, const float* __restrict__ attn_r,
    float* __restrict__ el, float* __restrict__ er, int n_nodes)
{
    const int idx = blockIdx.x * blockDim.x + threadIdx.x;   // n*HEADS + h
    if (idx >= n_nodes * HEADS) return;
    const int n = idx >> 2;
    const int h = idx & 3;
    const float* __restrict__ f  = feat   + (size_t)n * FEATD + h * HDIM;
    const float* __restrict__ al = attn_l + h * HDIM;
    const float* __restrict__ ar = attn_r + h * HDIM;
    float sl = 0.f, sr = 0.f;
#pragma unroll
    for (int d = 0; d < HDIM; ++d) {
        const float v = f[d];
        sl = fmaf(v, al[d], sl);
        sr = fmaf(v, ar[d], sr);
    }
    el[idx] = sl;
    er[idx] = sr;
}

// ---------------------------------------------------------------------------
// 3) row_ptr[n] = lower_bound(dst, n)  (dst is sorted)  for n in [0, N]
// ---------------------------------------------------------------------------
__global__ __launch_bounds__(256) void gat_rowptr(
    const int* __restrict__ dst, int* __restrict__ rowptr,
    int n_nodes, int n_edges)
{
    const int n = blockIdx.x * blockDim.x + threadIdx.x;
    if (n > n_nodes) return;
    int lo = 0, hi = n_edges;
    while (lo < hi) {
        const int mid = (lo + hi) >> 1;
        if (dst[mid] < n) lo = mid + 1; else hi = mid;
    }
    rowptr[n] = lo;
}

// ---------------------------------------------------------------------------
// 4) Per-destination-node edge softmax + weighted aggregation.
//    One wave32 per node (8 nodes / 256-thread block).
// ---------------------------------------------------------------------------
__global__ __launch_bounds__(256) void gat_aggregate(
    const float* __restrict__ feat,
    const float* __restrict__ el, const float* __restrict__ er,
    const int* __restrict__ src, const int* __restrict__ rowptr,
    float* __restrict__ out, int n_nodes)
{
    const int node = (blockIdx.x << 3) + (threadIdx.x >> 5);
    const int lane = threadIdx.x & 31;
    if (node >= n_nodes) return;

    const int beg = rowptr[node];
    const int end = rowptr[node + 1];
    const int fo  = lane << 2;                     // 4 output floats per lane

    const float4 ern = *(const float4*)(er + (size_t)node * HEADS);

    // --- pass 1: per-head max of leaky-relu logits over incoming edges ---
    float mx0 = -3.402823466e38f, mx1 = mx0, mx2 = mx0, mx3 = mx0;
    for (int i = beg + lane; i < end; i += 32) {
        const int s = src[i];
        const float4 e4 = *(const float4*)(el + (size_t)s * HEADS);
        float e;
        e = e4.x + ern.x; e = e > 0.f ? e : NEG_SLOPE * e; mx0 = fmaxf(mx0, e);
        e = e4.y + ern.y; e = e > 0.f ? e : NEG_SLOPE * e; mx1 = fmaxf(mx1, e);
        e = e4.z + ern.z; e = e > 0.f ? e : NEG_SLOPE * e; mx2 = fmaxf(mx2, e);
        e = e4.w + ern.w; e = e > 0.f ? e : NEG_SLOPE * e; mx3 = fmaxf(mx3, e);
    }
#pragma unroll
    for (int off = 16; off > 0; off >>= 1) {
        mx0 = fmaxf(mx0, __shfl_xor(mx0, off, 32));
        mx1 = fmaxf(mx1, __shfl_xor(mx1, off, 32));
        mx2 = fmaxf(mx2, __shfl_xor(mx2, off, 32));
        mx3 = fmaxf(mx3, __shfl_xor(mx3, off, 32));
    }

    // --- pass 2: per-head sum of exp(e - max) ---
    float s0 = 0.f, s1 = 0.f, s2 = 0.f, s3 = 0.f;
    for (int i = beg + lane; i < end; i += 32) {
        const int s = src[i];
        const float4 e4 = *(const float4*)(el + (size_t)s * HEADS);
        float e;
        e = e4.x + ern.x; e = e > 0.f ? e : NEG_SLOPE * e; s0 += __expf(e - mx0);
        e = e4.y + ern.y; e = e > 0.f ? e : NEG_SLOPE * e; s1 += __expf(e - mx1);
        e = e4.z + ern.z; e = e > 0.f ? e : NEG_SLOPE * e; s2 += __expf(e - mx2);
        e = e4.w + ern.w; e = e > 0.f ? e : NEG_SLOPE * e; s3 += __expf(e - mx3);
    }
#pragma unroll
    for (int off = 16; off > 0; off >>= 1) {
        s0 += __shfl_xor(s0, off, 32);
        s1 += __shfl_xor(s1, off, 32);
        s2 += __shfl_xor(s2, off, 32);
        s3 += __shfl_xor(s3, off, 32);
    }

    // --- pass 3: serial edges, parallel features (lane -> 4 floats of head lane/8)
    const int   myh  = lane >> 3;
    const float mymx = (myh == 0) ? mx0 : (myh == 1) ? mx1 : (myh == 2) ? mx2 : mx3;
    const float mysm = (myh == 0) ? s0  : (myh == 1) ? s1  : (myh == 2) ? s2  : s3;
    const float myer = (myh == 0) ? ern.x : (myh == 1) ? ern.y : (myh == 2) ? ern.z : ern.w;
    const float inv  = (end > beg) ? 1.0f / mysm : 0.0f;

    float a0 = 0.f, a1 = 0.f, a2 = 0.f, a3 = 0.f;
    for (int i = beg; i < end; ++i) {
        const int s = src[i];
        if (i + 1 < end)
            __builtin_prefetch(feat + (size_t)src[i + 1] * FEATD + fo, 0, 1);
        float e = el[(size_t)s * HEADS + myh] + myer;
        e = e > 0.f ? e : NEG_SLOPE * e;
        const float alpha = __expf(e - mymx) * inv;
        const float4 f4 = *(const float4*)(feat + (size_t)s * FEATD + fo);
        a0 = fmaf(alpha, f4.x, a0);
        a1 = fmaf(alpha, f4.y, a1);
        a2 = fmaf(alpha, f4.z, a2);
        a3 = fmaf(alpha, f4.w, a3);
    }
    float4 o; o.x = a0; o.y = a1; o.z = a2; o.w = a3;
    *(float4*)(out + (size_t)node * FEATD + fo) = o;
}

// ---------------------------------------------------------------------------
extern "C" void kernel_launch(void* const* d_in, const int* in_sizes, int n_in,
                              void* d_out, int out_size, void* d_ws, size_t ws_size,
                              hipStream_t stream)
{
    const float* x      = (const float*)d_in[0];
    const float* W      = (const float*)d_in[1];
    const float* attn_l = (const float*)d_in[2];
    const float* attn_r = (const float*)d_in[3];
    const int*   src    = (const int*)d_in[4];
    const int*   dst    = (const int*)d_in[5];
    float*       out    = (float*)d_out;

    const int  n_nodes = in_sizes[0] / FEATD;
    const int  n_edges = in_sizes[4];
    const long xn      = (long)n_nodes * FEATD;   // x element count
    const long wn      = (long)FEATD * FEATD;     // W element count

    // workspace layout (16B-aligned sections)
    char* ws = (char*)d_ws;
    size_t off = 0;
    auto alloc = [&](size_t bytes) { char* p = ws + off; off = (off + bytes + 15) & ~(size_t)15; return p; };
    float*    feat   = (float*)   alloc(sizeof(float) * xn);
    float*    el     = (float*)   alloc(sizeof(float) * n_nodes * HEADS);
    float*    er     = (float*)   alloc(sizeof(float) * n_nodes * HEADS);
    int*      rowptr = (int*)     alloc(sizeof(int) * (n_nodes + 1));
    _Float16* xh     = (_Float16*)alloc(sizeof(_Float16) * xn);
    _Float16* xl     = (_Float16*)alloc(sizeof(_Float16) * xn);
    _Float16* whi    = (_Float16*)alloc(sizeof(_Float16) * wn);
    _Float16* wlo    = (_Float16*)alloc(sizeof(_Float16) * wn);

    // 0) one-time f32 -> f16 hi/lo splits
    cvt_hilo<<<(int)((xn / 4 + 255) / 256), 256, 0, stream>>>(x, xh, xl, xn);
    cvt_hilo<<<(int)((wn / 4 + 255) / 256), 256, 0, stream>>>(W, whi, wlo, wn);

    // 1) WMMA projection
    gat_proj_wmma<<<(n_nodes + 15) / 16, 256, 0, stream>>>(xh, xl, whi, wlo, feat, n_nodes);

    // 2) attention scalars
    const int nh = n_nodes * HEADS;
    gat_el_er<<<(nh + 255) / 256, 256, 0, stream>>>(feat, attn_l, attn_r, el, er, n_nodes);

    // 3) CSR from sorted dst
    gat_rowptr<<<(n_nodes + 1 + 255) / 256, 256, 0, stream>>>(dst, rowptr, n_nodes, n_edges);

    // 4) edge softmax + gather-aggregate (HBM-bound phase)
    gat_aggregate<<<(n_nodes + 7) / 8, 256, 0, stream>>>(feat, el, er, src, rowptr, out, n_nodes);
}
